// SO3DeformableAttn_3410204033226
// MI455X (gfx1250) — matrix operand order repaired and checked by reading, not compiled, not run on hardware
//
#include <hip/hip_runtime.h>
#include <hip/hip_bf16.h>

// ---------------- problem constants ----------------
constexpr int BS = 4, NS = 2048, Fc = 256, HWp = 128, NH = 4, E = 64, G = 8;
constexpr int NQ = BS * NS;            // 8192 queries
constexpr int PLANE = Fc * HWp * HWp;  // elems per (batch) plane
constexpr int PIX = HWp * HWp;         // 16384 texels per plane

// ---------------- WMMA helpers (gfx1250, wave32) ----------------
typedef __attribute__((ext_vector_type(16))) __bf16 v16bf;
typedef __attribute__((ext_vector_type(8)))  float  v8f;

static __device__ __forceinline__ v8f wmma_bf16(v16bf a, v16bf b, v8f c) {
  // D = A(16x32 bf16) * B(32x16 bf16) + C(16x16 f32)
  return __builtin_amdgcn_wmma_f32_16x16x32_bf16(
      /*neg_a=*/false, a, /*neg_b=*/false, b,
      /*c_mod=*/(short)0, c, /*reuse_a=*/false, /*reuse_b=*/false);
}

// A tile: 16x32 bf16, row-major source with leading dim lda (elements).
static __device__ __forceinline__ v16bf load_A16x32(const __bf16* __restrict__ A,
                                                    int lda, int lane) {
  const __bf16* row = A + (lane & 15) * lda;
  int kb = (lane >> 4) * 8;
  v16bf a;
#pragma unroll
  for (int e = 0; e < 8; ++e) a[e] = row[kb + e];
#pragma unroll
  for (int e = 0; e < 8; ++e) a[8 + e] = row[16 + kb + e];
  return a;
}

// B fragment from pre-swizzled weights: one contiguous 32B load per lane.
static __device__ __forceinline__ v16bf load_B_packed(const __bf16* __restrict__ P,
                                                      int ks, int nt, int ntiles,
                                                      int lane) {
  const __bf16* p = P + (((size_t)(ks * ntiles + nt)) * 32 + lane) * 16;
  return *(const v16bf*)p;  // 32-byte aligned contiguous fragment
}

// ---------------- bilinear triplane sampler, stride-parameterized ----------------
// element address = plane + c*cs + (y*HW + x)*ps
//   channel-major (B,F,H,W): cs = HW*HW, ps = 1
//   pixel-major   (B,H,W,F): cs = 1,     ps = F    (coalesced across channels)
static __device__ __forceinline__ float bisample(const float* __restrict__ plane,
                                                 float u, float v, int c,
                                                 int cs, int ps) {
  const float mx = (float)(HWp - 1);
  float x = fminf(fmaxf((u + 1.0f) * 0.5f * mx, 0.0f), mx);
  float y = fminf(fmaxf((v + 1.0f) * 0.5f * mx, 0.0f), mx);
  float x0f = floorf(x), y0f = floorf(y);
  int x0 = (int)x0f, y0 = (int)y0f;
  int x1 = min(x0 + 1, HWp - 1), y1 = min(y0 + 1, HWp - 1);
  float wx = x - x0f, wy = y - y0f;
  const float* p = plane + (size_t)c * cs;
  float f00 = p[(size_t)(y0 * HWp + x0) * ps];
  float f01 = p[(size_t)(y0 * HWp + x1) * ps];
  float f10 = p[(size_t)(y1 * HWp + x0) * ps];
  float f11 = p[(size_t)(y1 * HWp + x1) * ps];
  return (f00 * (1.f - wx) + f01 * wx) * (1.f - wy) +
         (f10 * (1.f - wx) + f11 * wx) * wy;
}

static __device__ __forceinline__ float sample3(const float* __restrict__ pxz,
                                                const float* __restrict__ pxy,
                                                const float* __restrict__ pyz,
                                                float x, float y, float z, int c,
                                                int cs, int ps) {
  return bisample(pxz, x, z, c, cs, ps) + bisample(pxy, x, y, c, cs, ps) +
         bisample(pyz, y, z, c, cs, ps);
}

// ---------------- K-1: plane transpose (B,F,P) -> (B,P,F), LDS-tiled -----------
__global__ void k_transpose(const float* __restrict__ src, float* __restrict__ dst) {
  __shared__ float tile[32][33];
  int b = blockIdx.z;
  int p0 = blockIdx.x * 32;   // pixel tile
  int c0 = blockIdx.y * 32;   // channel tile
  const float* s = src + (size_t)b * PLANE;
  float* d = dst + (size_t)b * PLANE;
  for (int i = threadIdx.y; i < 32; i += 8)
    tile[i][threadIdx.x] = s[(size_t)(c0 + i) * PIX + p0 + threadIdx.x];
  __syncthreads();
  for (int i = threadIdx.y; i < 32; i += 8)
    d[(size_t)(p0 + i) * Fc + c0 + threadIdx.x] = tile[threadIdx.x][i];
}

// ---------------- K0: pack f32 KxN weight into bf16 WMMA B-fragment order ------
__global__ void k_pack(const float* __restrict__ W, __bf16* __restrict__ P, int N) {
  int i = blockIdx.x * 256 + threadIdx.x;
  int total = Fc * N;
  if (i >= total) return;
  int ntiles = N >> 4;
  int e = i & 15;
  int lane = (i >> 4) & 31;
  int tile = i >> 9;
  int nt = tile % ntiles;
  int ks = tile / ntiles;
  int k = ks * 32 + (lane >> 4) * 8 + (e & 7) + (e >> 3) * 16;
  int col = nt * 16 + (lane & 15);
  P[i] = (__bf16)W[(size_t)k * N + col];
}

// ---------------- K1: per-query feature sample + anchors ----------------
__global__ void k_feat(const float* __restrict__ qpos,
                       const float* __restrict__ cxz, const float* __restrict__ cxy,
                       const float* __restrict__ cyz, const float* __restrict__ grid,
                       float* __restrict__ featv, float* __restrict__ anchor,
                       int cs, int ps) {
  int q = blockIdx.x;
  int b = q / NS;
  int t = threadIdx.x;                // channel 0..255
  const float* qp = qpos + (size_t)q * 7;
  float px = qp[0], py = qp[1], pz = qp[2];
  const float* pxz = cxz + (size_t)b * PLANE;
  const float* pxy = cxy + (size_t)b * PLANE;
  const float* pyz = cyz + (size_t)b * PLANE;
  featv[(size_t)q * Fc + t] = sample3(pxz, pxy, pyz, px, py, pz, t, cs, ps);

  if (t < G) {
    float r = qp[3], qi = qp[4], qj = qp[5], qk = qp[6];
    float two_s = 2.0f / (r * r + qi * qi + qj * qj + qk * qk);
    float m00 = 1.f - two_s * (qj * qj + qk * qk);
    float m01 = two_s * (qi * qj - qk * r);
    float m02 = two_s * (qi * qk + qj * r);
    float m10 = two_s * (qi * qj + qk * r);
    float m11 = 1.f - two_s * (qi * qi + qk * qk);
    float m12 = two_s * (qj * qk - qi * r);
    float m20 = two_s * (qi * qk - qj * r);
    float m21 = two_s * (qj * qk + qi * r);
    float m22 = 1.f - two_s * (qi * qi + qj * qj);
    float gx = grid[t * 3 + 0], gy = grid[t * 3 + 1], gz = grid[t * 3 + 2];
    float* a = anchor + ((size_t)q * G + t) * 3;
    a[0] = px + m00 * gx + m01 * gy + m02 * gz;
    a[1] = py + m10 * gx + m11 * gy + m12 * gz;
    a[2] = pz + m20 * gx + m21 * gy + m22 * gz;
  }
}

// ---------------- K2: q = (feature @ W_q + b_q) * sqrt(E), WMMA ----------------
__global__ void k_qproj(const float* __restrict__ featv, const __bf16* __restrict__ Wqp,
                        const float* __restrict__ bq, float* __restrict__ qv) {
  __shared__ __bf16 At[16 * Fc];
  int m0 = blockIdx.x * 16;
  for (int idx = threadIdx.x; idx < 16 * Fc; idx += blockDim.x)
    At[idx] = (__bf16)featv[(size_t)(m0 + idx / Fc) * Fc + (idx % Fc)];
  __syncthreads();
  int wave = threadIdx.x >> 5, lane = threadIdx.x & 31;
  int nt = wave;                      // 4 waves cover N=64 (4 n-tiles)
  v8f acc = {};
#pragma unroll
  for (int ks = 0; ks < Fc / 32; ++ks) {
    v16bf a = load_A16x32(At + ks * 32, Fc, lane);
    v16bf b = load_B_packed(Wqp, ks, nt, E / 16, lane);
    acc = wmma_bf16(a, b, acc);
  }
  int col = nt * 16 + (lane & 15);
  int gg = lane >> 4;
#pragma unroll
  for (int j = 0; j < 8; ++j) {
    int row = m0 + gg * 8 + j;
    qv[(size_t)row * E + col] = (acc[j] + bq[col]) * 8.0f;  // sqrt(64)
  }
}

// ---------------- K3: anchor features -> ctx -> offsets -> sample points ----------
__global__ void k_afctx(const float* __restrict__ cxz, const float* __restrict__ cxy,
                        const float* __restrict__ cyz, const float* __restrict__ anchor,
                        const float* __restrict__ Woc, const float* __restrict__ boc,
                        const float* __restrict__ Woff, float* __restrict__ sp,
                        int cs, int ps) {
  int q = blockIdx.x;
  int b = q / NS;
  int t = threadIdx.x;                // 256 threads
  __shared__ float af[G][Fc];
  __shared__ float meanv[Fc];
  __shared__ float ctxb[Fc];
  __shared__ float anc[G * 3];
  if (t < G * 3) anc[t] = anchor[(size_t)q * G * 3 + t];
  __syncthreads();
  const float* pxz = cxz + (size_t)b * PLANE;
  const float* pxy = cxy + (size_t)b * PLANE;
  const float* pyz = cyz + (size_t)b * PLANE;
#pragma unroll
  for (int p = 0; p < G; ++p)
    af[p][t] = sample3(pxz, pxy, pyz, anc[p * 3], anc[p * 3 + 1], anc[p * 3 + 2],
                       t, cs, ps);
  __syncthreads();
  float m = 0.f;
#pragma unroll
  for (int p = 0; p < G; ++p) m += af[p][t];
  meanv[t] = m * (1.0f / G);
  __syncthreads();
  {
    int h = t >> 6, e = t & 63;
    float s = boc[e];
    for (int e2 = 0; e2 < E; ++e2) s += meanv[h * E + e2] * Woc[e2 * E + e];
    ctxb[t] = fmaxf(s, 0.f);
  }
  __syncthreads();
  if (t < 96) {
    int pair = t / 3, d = t % 3;
    int g = pair >> 2, h = pair & 3;
    float s = 0.f;
    for (int e = 0; e < E; ++e)
      s += (ctxb[h * E + e] + af[g][h * E + e]) * Woff[e * 3 + d];
    sp[((size_t)q * 32 + pair) * 3 + d] = anc[g * 3 + d] + s;
  }
}

// ---------------- K4: fused sample sf -> K/V (WMMA) -> attention ----------------
__global__ void k_attn(const float* __restrict__ cxz, const float* __restrict__ cxy,
                       const float* __restrict__ cyz, const float* __restrict__ sp,
                       const float* __restrict__ qv,
                       const __bf16* __restrict__ Wkp, const __bf16* __restrict__ Wvp,
                       const float* __restrict__ bk, const float* __restrict__ bv,
                       float* __restrict__ attnout, int cs, int ps) {
  int q = blockIdx.x;
  int b = q / NS;
  int t = threadIdx.x;                // 256 threads = 8 waves
  __shared__ __bf16 sfb[32 * Fc];     // 16 KB
  __shared__ float kk[32 * E];        // 8 KB
  __shared__ float vv[32 * E];        // 8 KB
  __shared__ float qs[E];
  __shared__ float attnw[32];
  __shared__ float spl[96];
  if (t < E) qs[t] = qv[(size_t)q * E + t];
  if (t < 96) spl[t] = sp[(size_t)q * 96 + t];
  __syncthreads();
  const float* pxz = cxz + (size_t)b * PLANE;
  const float* pxy = cxy + (size_t)b * PLANE;
  const float* pyz = cyz + (size_t)b * PLANE;
#pragma unroll 4
  for (int p = 0; p < 32; ++p) {
    float x = spl[p * 3], y = spl[p * 3 + 1], z = spl[p * 3 + 2];
    sfb[p * Fc + t] = (__bf16)sample3(pxz, pxy, pyz, x, y, z, t, cs, ps);
  }
  __syncthreads();
  // K = sf@Wk + bk, V = sf@Wv + bv; M=32 (2 tiles), N=64 (4 tiles), K=256 (8 steps)
  {
    int wave = t >> 5, lane = t & 31;
    int mt = wave & 1, nt = wave >> 1;
    v8f ak = {}, av = {};
#pragma unroll
    for (int ks = 0; ks < Fc / 32; ++ks) {
      v16bf a   = load_A16x32(sfb + mt * 16 * Fc + ks * 32, Fc, lane);
      v16bf bkt = load_B_packed(Wkp, ks, nt, E / 16, lane);
      v16bf bvt = load_B_packed(Wvp, ks, nt, E / 16, lane);
      ak = wmma_bf16(a, bkt, ak);
      av = wmma_bf16(a, bvt, av);
    }
    int col = nt * 16 + (lane & 15);
    int gg = lane >> 4;
#pragma unroll
    for (int j = 0; j < 8; ++j) {
      int row = mt * 16 + gg * 8 + j;
      kk[row * E + col] = ak[j] + bk[col];
      vv[row * E + col] = av[j] + bv[col];
    }
  }
  __syncthreads();
  // sim + softmax over g (wave 0; lane = pair = g*4+h, h = lane&3)
  if (t < 32) {
    float s = 0.f;
#pragma unroll
    for (int e = 0; e < E; ++e) s += qs[e] * kk[t * E + e];
    float mx = s;
    mx = fmaxf(mx, __shfl_xor(mx, 4, 32));
    mx = fmaxf(mx, __shfl_xor(mx, 8, 32));
    mx = fmaxf(mx, __shfl_xor(mx, 16, 32));
    float ex = __expf(s - mx);
    float sum = ex;
    sum += __shfl_xor(sum, 4, 32);
    sum += __shfl_xor(sum, 8, 32);
    sum += __shfl_xor(sum, 16, 32);
    attnw[t] = ex / sum;
  }
  __syncthreads();
  {
    int h = t >> 6, e = t & 63;
    float o = 0.f;
#pragma unroll
    for (int g = 0; g < G; ++g) o += attnw[g * NH + h] * vv[(g * NH + h) * E + e];
    attnout[(size_t)q * Fc + t] = o;
  }
}

// ---------------- K5: final out = attn @ W_out + b_out + feature (WMMA) ----------
__global__ void k_out(const float* __restrict__ attnout, const __bf16* __restrict__ Wop,
                      const float* __restrict__ bout, const float* __restrict__ featv,
                      float* __restrict__ out) {
  __shared__ __bf16 At[16 * Fc];
  int m0 = blockIdx.x * 16;
  for (int idx = threadIdx.x; idx < 16 * Fc; idx += blockDim.x)
    At[idx] = (__bf16)attnout[(size_t)(m0 + idx / Fc) * Fc + (idx % Fc)];
  __syncthreads();
  int wave = threadIdx.x >> 5, lane = threadIdx.x & 31;
  int nt0 = wave * 2;                 // 8 waves x 2 n-tiles cover N=256
  v8f a0 = {}, a1 = {};
#pragma unroll
  for (int ks = 0; ks < Fc / 32; ++ks) {
    v16bf a  = load_A16x32(At + ks * 32, Fc, lane);
    v16bf b0 = load_B_packed(Wop, ks, nt0, Fc / 16, lane);
    v16bf b1 = load_B_packed(Wop, ks, nt0 + 1, Fc / 16, lane);
    a0 = wmma_bf16(a, b0, a0);
    a1 = wmma_bf16(a, b1, a1);
  }
  int gg = lane >> 4, c0 = nt0 * 16 + (lane & 15);
#pragma unroll
  for (int j = 0; j < 8; ++j) {
    int row = m0 + gg * 8 + j;
    out[(size_t)row * Fc + c0]      = a0[j] + bout[c0]      + featv[(size_t)row * Fc + c0];
    out[(size_t)row * Fc + c0 + 16] = a1[j] + bout[c0 + 16] + featv[(size_t)row * Fc + c0 + 16];
  }
}

// ---------------- launcher ----------------
extern "C" void kernel_launch(void* const* d_in, const int* in_sizes, int n_in,
                              void* d_out, int out_size, void* d_ws, size_t ws_size,
                              hipStream_t stream) {
  const float* query_pos = (const float*)d_in[0];
  const float* c_xz  = (const float*)d_in[1];
  const float* c_xy  = (const float*)d_in[2];
  const float* c_yz  = (const float*)d_in[3];
  const float* grid  = (const float*)d_in[4];
  const float* W_q   = (const float*)d_in[5];
  const float* b_q   = (const float*)d_in[6];
  const float* W_k   = (const float*)d_in[7];
  const float* b_k   = (const float*)d_in[8];
  const float* W_v   = (const float*)d_in[9];
  const float* b_v   = (const float*)d_in[10];
  const float* W_oc  = (const float*)d_in[11];
  const float* b_oc  = (const float*)d_in[12];
  const float* W_off = (const float*)d_in[13];
  const float* W_out = (const float*)d_in[14];
  const float* b_out = (const float*)d_in[15];
  float* out = (float*)d_out;

  char* p = (char*)d_ws;
  char* pend = (char*)d_ws + ws_size;
  auto take = [&](size_t bytes) { char* r = p; p += (bytes + 255) & ~size_t(255); return r; };
  float*  featv   = (float*)take((size_t)NQ * Fc * 4);
  float*  anchor  = (float*)take((size_t)NQ * G * 3 * 4);
  float*  sp      = (float*)take((size_t)NQ * 32 * 3 * 4);
  float*  qvv     = (float*)take((size_t)NQ * E * 4);
  float*  attnout = (float*)take((size_t)NQ * Fc * 4);
  __bf16* Wqp = (__bf16*)take((size_t)Fc * E * 2);
  __bf16* Wkp = (__bf16*)take((size_t)Fc * E * 2);
  __bf16* Wvp = (__bf16*)take((size_t)Fc * E * 2);
  __bf16* Wop = (__bf16*)take((size_t)Fc * Fc * 2);

  // Optional pixel-major transposed planes (coalesced channel sampling):
  // needs 3 * BS*F*HW*HW*4 = 201 MB extra workspace.
  const size_t planeBytes = (size_t)BS * PLANE * 4;
  bool useT = (pend - p) >= (ptrdiff_t)(3 * (planeBytes + 256));
  const float *sxz = c_xz, *sxy = c_xy, *syz = c_yz;
  int cs = PIX, ps = 1;               // channel-major fallback
  if (useT) {
    float* txz = (float*)take(planeBytes);
    float* txy = (float*)take(planeBytes);
    float* tyz = (float*)take(planeBytes);
    dim3 tg(PIX / 32, Fc / 32, BS), tb(32, 8);
    k_transpose<<<tg, tb, 0, stream>>>(c_xz, txz);
    k_transpose<<<tg, tb, 0, stream>>>(c_xy, txy);
    k_transpose<<<tg, tb, 0, stream>>>(c_yz, tyz);
    sxz = txz; sxy = txy; syz = tyz;
    cs = 1; ps = Fc;                  // pixel-major: channels contiguous
  }

  k_pack<<<(Fc * E + 255) / 256, 256, 0, stream>>>(W_q, Wqp, E);
  k_pack<<<(Fc * E + 255) / 256, 256, 0, stream>>>(W_k, Wkp, E);
  k_pack<<<(Fc * E + 255) / 256, 256, 0, stream>>>(W_v, Wvp, E);
  k_pack<<<(Fc * Fc + 255) / 256, 256, 0, stream>>>(W_out, Wop, Fc);
  k_feat<<<NQ, 256, 0, stream>>>(query_pos, sxz, sxy, syz, grid, featv, anchor, cs, ps);
  k_qproj<<<NQ / 16, 128, 0, stream>>>(featv, Wqp, b_q, qvv);
  k_afctx<<<NQ, 256, 0, stream>>>(sxz, sxy, syz, anchor, W_oc, b_oc, W_off, sp, cs, ps);
  k_attn<<<NQ, 256, 0, stream>>>(sxz, sxy, syz, sp, qvv, Wkp, Wvp, b_k, b_v, attnout,
                                 cs, ps);
  k_out<<<NQ / 16, 256, 0, stream>>>(attnout, Wop, b_out, featv, out);
}
